// MultiHeadAttentionQuantum_65481071405845
// MI455X (gfx1250) — matrix operand order repaired and checked by reading, not compiled, not run on hardware
//
#include <hip/hip_runtime.h>
#include <hip/hip_bf16.h>

// ---------------------------------------------------------------------------
// Quantum-flavored single-head attention, fully on CDNA5 WMMA (bf16 in, f32 acc)
//   B=8, T=2048, E=256, NW=8  ->  N = B*T = 16384 rows
// Pipeline:
//   (1) f32 -> bf16 convert (x, Wq, Wk, Wv, Wo)
//   (2) proj+cos:  q_q = cos(x@Wq^T + th), k_q likewise (row-major),
//                  v_q = cos(x@Wv^T + th) stored TRANSPOSED [B][E][T]
//   (3) flash attention (online softmax), ctx stored bf16 row-major
//   (4) out = ctx @ Wo^T  (fp32 output)
// ---------------------------------------------------------------------------

typedef __attribute__((ext_vector_type(16))) __bf16 v16bf;
typedef __attribute__((ext_vector_type(8)))  __bf16 v8bf;
typedef __attribute__((ext_vector_type(8)))  float  v8f;

#define E_DIM 256
#define T_DIM 2048
#define B_DIM 8
#define N_ROWS (B_DIM * T_DIM)

// ---- WMMA fragment helpers (contiguous k = 16h+s in both A and B, wave32) ----------

// A-matrix 16x32 bf16: lane L holds row M = L&15.
//   lanes 0-15 : K 0..15,  lanes 16-31: K 16..31 (contiguous, matches B)
__device__ __forceinline__ v16bf load_a_frag(const __bf16* row_base, int kk, int lane) {
    const int c0 = (lane >> 4) << 4;                 // 0 or 16
    v8bf lo = *(const v8bf*)(row_base + kk + c0);
    v8bf hi = *(const v8bf*)(row_base + kk + c0 + 8);
    v16bf a;
#pragma unroll
    for (int i = 0; i < 8; ++i) { a[i] = lo[i]; a[i + 8] = hi[i]; }
    return a;
}

// B-matrix 32x16 bf16: lane L holds column N = L&15.
//   lanes 0-15 hold K=0..15, lanes 16-31 hold K=16..31 (VGPR v = K 2v,2v+1).
// Caller passes pointer to the 16 contiguous K-elements for this lane.
__device__ __forceinline__ v16bf load_b_frag(const __bf16* p) {
    v8bf lo = *(const v8bf*)(p);
    v8bf hi = *(const v8bf*)(p + 8);
    v16bf b;
#pragma unroll
    for (int i = 0; i < 8; ++i) { b[i] = lo[i]; b[i + 8] = hi[i]; }
    return b;
}

__device__ __forceinline__ v8f wmma_bf16(v16bf a, v16bf b, v8f c) {
    // (neg_a, A, neg_b, B, c_mod, C, reuse_a, reuse_b)
    return __builtin_amdgcn_wmma_f32_16x16x32_bf16(false, a, false, b, (short)0, c,
                                                   false, false);
}

// ---- (1) fp32 -> bf16 ------------------------------------------------------

__global__ void cvt_f32_bf16_kernel(const float* __restrict__ in,
                                    __bf16* __restrict__ out, int n) {
    int i = blockIdx.x * blockDim.x + threadIdx.x;
    if (i < n) out[i] = (__bf16)in[i];
}

// ---- (2) projection GEMM + cos(z + theta[e%8]) -----------------------------
// out[n_row, e] = cos( sum_k X[n_row,k] * W[e,k] + theta[e%8] )
// transpose==0: row-major [N_ROWS, E]; transpose==1: per-batch [E, T]

__global__ void proj_cos_kernel(const __bf16* __restrict__ X,
                                const __bf16* __restrict__ W,
                                const float* __restrict__ theta,
                                __bf16* __restrict__ out, int transpose) {
    const int mt = blockIdx.x;              // (N_ROWS/16) row tiles
    const int nt = blockIdx.y;              // (E/16) col tiles
    const int lane = threadIdx.x;
    const int lo16 = lane & 15, hi = lane >> 4;
    const int kbase = hi << 4;

    const __bf16* xrow = X + (size_t)(mt * 16 + lo16) * E_DIM;
    const __bf16* wrow = W + (size_t)(nt * 16 + lo16) * E_DIM;

    v8f c = {};
#pragma unroll
    for (int kk = 0; kk < E_DIM; kk += 32) {
        v16bf a = load_a_frag(xrow, kk, lane);
        v16bf b = load_b_frag(wrow + kk + kbase);
        c = wmma_bf16(a, b, c);
    }

    const float th = theta[lane & 7];       // e = nt*16 + lo16, 16%8==0
    const int e = nt * 16 + lo16;
#pragma unroll
    for (int r = 0; r < 8; ++r) {
        const int m = mt * 16 + r + 8 * hi; // global row in [0, N_ROWS)
        const float val = __cosf(c[r] + th);   // hardware v_cos_f32, no libm slow path
        if (transpose) {
            const int bb = m / T_DIM, t = m % T_DIM;
            out[((size_t)bb * E_DIM + e) * T_DIM + t] = (__bf16)val;
        } else {
            out[(size_t)m * E_DIM + e] = (__bf16)val;
        }
    }
}

// ---- (3) flash attention ---------------------------------------------------
// One wave handles 16 queries; keys streamed in chunks of 32.

__global__ void flash_attn_kernel(const __bf16* __restrict__ Q,   // [N_ROWS,E]
                                  const __bf16* __restrict__ K,   // [N_ROWS,E]
                                  const __bf16* __restrict__ Vt,  // [B][E][T]
                                  __bf16* __restrict__ ctx) {     // [N_ROWS,E]
    const int qt = blockIdx.x;              // T/16 query tiles
    const int bb = blockIdx.y;              // batch
    const int lane = threadIdx.x;
    const int lo16 = lane & 15, hi = lane >> 4;
    const int kbase = hi << 4;
    const float scale = 0.35355339059327373f;   // 1/sqrt(NW=8)

    __shared__ __bf16 Plds[16][32];         // P tile: 16 queries x 32 keys

    // Preload all Q A-fragments (row = this lane's query row)
    const __bf16* qrow = Q + ((size_t)bb * T_DIM + qt * 16 + lo16) * E_DIM;
    v16bf qfrag[8];
#pragma unroll
    for (int ks = 0; ks < 8; ++ks) qfrag[ks] = load_a_frag(qrow, ks * 32, lane);

    v8f O[16];
#pragma unroll
    for (int i = 0; i < 16; ++i) O[i] = (v8f){};
    float mrow[8], lrow[8];
#pragma unroll
    for (int r = 0; r < 8; ++r) { mrow[r] = -3.0e38f; lrow[r] = 0.0f; }

    const __bf16* vtb = Vt + (size_t)bb * E_DIM * T_DIM;

    for (int tc = 0; tc < T_DIM; tc += 32) {
        // ---- scores: two 16x16 tiles (keys tc..tc+15, tc+16..tc+31)
        v8f S0 = {}, S1 = {};
        const __bf16* k0 = K + ((size_t)bb * T_DIM + tc + lo16) * E_DIM + kbase;
        const __bf16* k1 = k0 + (size_t)16 * E_DIM;
        __builtin_prefetch(k0 + (size_t)32 * E_DIM, 0, 1);      // next K chunk
        __builtin_prefetch(vtb + tc + 32 + kbase + (size_t)lo16 * T_DIM, 0, 1); // next V chunk
#pragma unroll
        for (int ks = 0; ks < 8; ++ks) {
            v16bf bf0 = load_b_frag(k0 + ks * 32);
            v16bf bf1 = load_b_frag(k1 + ks * 32);
            S0 = wmma_bf16(qfrag[ks], bf0, S0);
            S1 = wmma_bf16(qfrag[ks], bf1, S1);
        }

        // ---- online softmax (rows split across 16-lane halves like C-frag)
#pragma unroll
        for (int r = 0; r < 8; ++r) {
            float s0 = S0[r] * scale, s1 = S1[r] * scale;
            float mx = fmaxf(s0, s1);
#pragma unroll
            for (int msk = 8; msk >= 1; msk >>= 1)
                mx = fmaxf(mx, __shfl_xor(mx, msk, 16));
            const float newm = fmaxf(mrow[r], mx);
            const float corr = __expf(mrow[r] - newm);
            const float p0 = __expf(s0 - newm);
            const float p1 = __expf(s1 - newm);
            float ps = p0 + p1;
#pragma unroll
            for (int msk = 8; msk >= 1; msk >>= 1)
                ps += __shfl_xor(ps, msk, 16);
            lrow[r] = lrow[r] * corr + ps;
            mrow[r] = newm;
#pragma unroll
            for (int et = 0; et < 16; ++et) O[et][r] *= corr;
            const int prow = r + 8 * hi;
            Plds[prow][lo16]      = (__bf16)p0;
            Plds[prow][16 + lo16] = (__bf16)p1;
        }
        __syncthreads();

        // ---- P (16x32) back as A-fragment, then PV accumulate over 16 e-tiles
        v16bf pfrag = load_a_frag(&Plds[lo16][0], 0, lane);
        const __bf16* vbase = vtb + tc + kbase;
#pragma unroll
        for (int et = 0; et < 16; ++et) {
            v16bf bf = load_b_frag(vbase + (size_t)(et * 16 + lo16) * T_DIM);
            O[et] = wmma_bf16(pfrag, bf, O[et]);
        }
        __syncthreads();
    }

    // ---- normalize, store ctx as bf16 row-major
#pragma unroll
    for (int r = 0; r < 8; ++r) {
        const float inv = 1.0f / lrow[r];
        const size_t row = (size_t)bb * T_DIM + qt * 16 + r + 8 * hi;
#pragma unroll
        for (int et = 0; et < 16; ++et)
            ctx[row * E_DIM + et * 16 + lo16] = (__bf16)(O[et][r] * inv);
    }
}

// ---- (4) output projection: out = ctx @ Wo^T (fp32 out) --------------------

__global__ void out_proj_kernel(const __bf16* __restrict__ Cx,
                                const __bf16* __restrict__ Wo,
                                float* __restrict__ out) {
    const int mt = blockIdx.x;
    const int nt = blockIdx.y;
    const int lane = threadIdx.x;
    const int lo16 = lane & 15, hi = lane >> 4;
    const int kbase = hi << 4;

    const __bf16* crow = Cx + (size_t)(mt * 16 + lo16) * E_DIM;
    const __bf16* wrow = Wo + (size_t)(nt * 16 + lo16) * E_DIM;

    v8f c = {};
#pragma unroll
    for (int kk = 0; kk < E_DIM; kk += 32) {
        v16bf a = load_a_frag(crow, kk, lane);
        v16bf b = load_b_frag(wrow + kk + kbase);
        c = wmma_bf16(a, b, c);
    }
    const int e = nt * 16 + lo16;
#pragma unroll
    for (int r = 0; r < 8; ++r) {
        const int m = mt * 16 + r + 8 * hi;
        out[(size_t)m * E_DIM + e] = c[r];
    }
}

// ---------------------------------------------------------------------------

extern "C" void kernel_launch(void* const* d_in, const int* in_sizes, int n_in,
                              void* d_out, int out_size, void* d_ws, size_t ws_size,
                              hipStream_t stream) {
    (void)in_sizes; (void)n_in; (void)out_size; (void)ws_size;

    const float* x     = (const float*)d_in[0];
    const float* Wq    = (const float*)d_in[1];
    const float* Wk    = (const float*)d_in[2];
    const float* Wv    = (const float*)d_in[3];
    const float* Wo    = (const float*)d_in[4];
    const float* theta = (const float*)d_in[5];
    float* out = (float*)d_out;

    // workspace layout (bytes), total ~40.5 MB
    char* ws = (char*)d_ws;
    const size_t xe  = (size_t)N_ROWS * E_DIM;     // 4,194,304 elems
    const size_t we  = (size_t)E_DIM * E_DIM;      // 65,536 elems
    __bf16* xb   = (__bf16*)(ws);
    __bf16* wqb  = (__bf16*)(ws + 2 * xe);
    __bf16* wkb  = (__bf16*)(ws + 2 * xe + 2 * we);
    __bf16* wvb  = (__bf16*)(ws + 2 * xe + 4 * we);
    __bf16* wob  = (__bf16*)(ws + 2 * xe + 6 * we);
    __bf16* qb   = (__bf16*)(ws + 2 * xe + 8 * we);
    __bf16* kb   = (__bf16*)(ws + 4 * xe + 8 * we);
    __bf16* vtb  = (__bf16*)(ws + 6 * xe + 8 * we);
    __bf16* ctxb = (__bf16*)(ws + 8 * xe + 8 * we);

    // (1) convert fp32 -> bf16
    cvt_f32_bf16_kernel<<<(int)((xe + 255) / 256), 256, 0, stream>>>(x,  xb,  (int)xe);
    cvt_f32_bf16_kernel<<<(int)((we + 255) / 256), 256, 0, stream>>>(Wq, wqb, (int)we);
    cvt_f32_bf16_kernel<<<(int)((we + 255) / 256), 256, 0, stream>>>(Wk, wkb, (int)we);
    cvt_f32_bf16_kernel<<<(int)((we + 255) / 256), 256, 0, stream>>>(Wv, wvb, (int)we);
    cvt_f32_bf16_kernel<<<(int)((we + 255) / 256), 256, 0, stream>>>(Wo, wob, (int)we);

    // (2) projections + quantum cos
    dim3 pgrid(N_ROWS / 16, E_DIM / 16);
    proj_cos_kernel<<<pgrid, 32, 0, stream>>>(xb, wqb, theta, qb,  0);
    proj_cos_kernel<<<pgrid, 32, 0, stream>>>(xb, wkb, theta, kb,  0);
    proj_cos_kernel<<<pgrid, 32, 0, stream>>>(xb, wvb, theta, vtb, 1);

    // (3) flash attention
    dim3 agrid(T_DIM / 16, B_DIM);
    flash_attn_kernel<<<agrid, 32, 0, stream>>>(qb, kb, vtb, ctxb);

    // (4) output projection
    out_proj_kernel<<<pgrid, 32, 0, stream>>>(ctxb, wob, out);
}
